// Decoder_34282428957170
// MI455X (gfx1250) — compile-verified, matrix operand order checked by reading, not run on hardware
//
#include <hip/hip_runtime.h>

// ---------------------------------------------------------------------------
// Types
// ---------------------------------------------------------------------------
typedef __attribute__((ext_vector_type(16))) __bf16         v16bf;
typedef __attribute__((ext_vector_type(8)))  float          v8f;
typedef unsigned short u16;
typedef __attribute__((ext_vector_type(8)))  unsigned short u16x8;
typedef __attribute__((ext_vector_type(16))) unsigned short u16x16;

__device__ __forceinline__ u16 f2bf(float f) {
  unsigned u = __builtin_bit_cast(unsigned, f);
  u += 0x7FFFu + ((u >> 16) & 1u);          // round-to-nearest-even
  return (u16)(u >> 16);
}
__device__ __forceinline__ float sigm(float x) { return 1.0f / (1.0f + __expf(-x)); }

// Problem constants
#define Bc  256
#define Tc  64
#define Sc  256
#define Ec  256
#define Hc  1024
#define Hec 1024
#define Ac  128
#define Vc  256
#define K1c 2304   /* E + He + H */
#define K2c 2048   /* H + H     */

// ---------------------------------------------------------------------------
// bf16 WMMA GEMM:  C[M,N](f32) = A[M,K](f32, converted on stage) * B[K,N](bf16) + bias[N]
// WG = 256 thr = 8 wave32 (2x4), tile 64x128, K-step 32, 2x2 WMMA frags/wave.
// ---------------------------------------------------------------------------
#define BM 64
#define BN 128
#define BK 32
#define LP 8   // LDS row pad (shorts); row stride 80B keeps 16B alignment

__global__ __launch_bounds__(256) void gemm_bf16_kernel(
    const float* __restrict__ A, int lda,
    const u16*   __restrict__ B,            // [K,N] bf16 row-major
    const float* __restrict__ bias,
    float* __restrict__ C, int ldc,
    int M, int N, int K)
{
  __shared__ u16 ldsA[BM][BK + LP];
  __shared__ u16 ldsB[BN][BK + LP];         // transposed: [n][k]

  const int tid  = threadIdx.x;
  const int lane = tid & 31;
  const int wave = tid >> 5;
  const int wm   = wave >> 2;               // 0..1
  const int wn   = wave & 3;                // 0..3
  const int half = lane >> 4;               // K-half select
  const int q16  = lane & 15;

  const int m0g = blockIdx.y * BM;
  const int n0g = blockIdx.x * BN;

  v8f acc[2][2] = {};

  // staging assignments
  const int am = tid >> 2;                  // 0..63 rows of A tile
  const int ak = (tid & 3) * 8;             // 0/8/16/24
  const int bk = tid >> 3;                  // 0..31 rows of B tile
  const int bn = (tid & 7) * 16;            // 0..112

  for (int k0 = 0; k0 < K; k0 += BK) {
    __syncthreads();
    { // A tile: 8 fp32 -> bf16 into LDS
      const float* src = A + (size_t)(m0g + am) * lda + (k0 + ak);
      float4 f0 = *(const float4*)(src);
      float4 f1 = *(const float4*)(src + 4);
      u16* d = &ldsA[am][ak];
      d[0] = f2bf(f0.x); d[1] = f2bf(f0.y); d[2] = f2bf(f0.z); d[3] = f2bf(f0.w);
      d[4] = f2bf(f1.x); d[5] = f2bf(f1.y); d[6] = f2bf(f1.z); d[7] = f2bf(f1.w);
    }
    { // B tile: 16 bf16, store transposed
      const u16* src = B + (size_t)(k0 + bk) * N + (n0g + bn);
      u16x8 b0 = *(const u16x8*)(src);
      u16x8 b1 = *(const u16x8*)(src + 8);
#pragma unroll
      for (int i = 0; i < 8; ++i) ldsB[bn + i][bk]     = b0[i];
#pragma unroll
      for (int i = 0; i < 8; ++i) ldsB[bn + 8 + i][bk] = b1[i];
    }
    __syncthreads();

    // Build fragments per documented 16-bit layout:
    // lane half=0: halves 0-7 => K 0-7, 8-15 => K 16-23 ; half=1: +8
    u16x16 afr[2], bfr[2];
#pragma unroll
    for (int mt = 0; mt < 2; ++mt) {
      const int m = wm * 32 + mt * 16 + q16;
      u16x8 lo = *(const u16x8*)&ldsA[m][8 * half];
      u16x8 hi = *(const u16x8*)&ldsA[m][8 * half + 16];
#pragma unroll
      for (int i = 0; i < 8; ++i) { afr[mt][i] = lo[i]; afr[mt][i + 8] = hi[i]; }
    }
#pragma unroll
    for (int nt = 0; nt < 2; ++nt) {
      const int n = wn * 32 + nt * 16 + q16;
      u16x8 lo = *(const u16x8*)&ldsB[n][8 * half];
      u16x8 hi = *(const u16x8*)&ldsB[n][8 * half + 16];
#pragma unroll
      for (int i = 0; i < 8; ++i) { bfr[nt][i] = lo[i]; bfr[nt][i + 8] = hi[i]; }
    }

#pragma unroll
    for (int mt = 0; mt < 2; ++mt)
#pragma unroll
      for (int nt = 0; nt < 2; ++nt)
        acc[mt][nt] = __builtin_amdgcn_wmma_f32_16x16x32_bf16(
            false, __builtin_bit_cast(v16bf, afr[mt]),
            false, __builtin_bit_cast(v16bf, bfr[nt]),
            (short)0, acc[mt][nt], false, false);
  }

  // epilogue: C/D layout — lane holds col n=q16(+16 grp), VGPR r -> row r+8*half
#pragma unroll
  for (int mt = 0; mt < 2; ++mt)
#pragma unroll
    for (int nt = 0; nt < 2; ++nt) {
      const int n  = n0g + wn * 32 + nt * 16 + q16;
      const float bv = bias[n];
#pragma unroll
      for (int r = 0; r < 8; ++r) {
        const int m = m0g + wm * 32 + mt * 16 + r + 8 * half;
        C[(size_t)m * ldc + n] = acc[mt][nt][r] + bv;
      }
    }
}

// ---------------------------------------------------------------------------
// Attention: per-batch softmax + context; also gathers embedding row.
// One WG (256 thr) per batch element.
// ---------------------------------------------------------------------------
__global__ __launch_bounds__(256) void attention_kernel(
    const float* __restrict__ q,     // [B,A]
    const float* __restrict__ encp,  // [B,S,A]
    const float* __restrict__ enc,   // [B,S,He]
    const float* __restrict__ Va,    // [A]
    const float* __restrict__ bV,    // [1]
    const float* __restrict__ emb,   // [V,E]
    const int*   __restrict__ x,     // [B,T]
    int t,
    float* __restrict__ act1)        // [B, K1] rows: [emb | ctx | h1]
{
  const int b   = blockIdx.x;
  const int tid = threadIdx.x;
  __shared__ float qv[Ac];
  __shared__ float sc[Sc];
  __shared__ float red[Sc];

  if (tid < Ac) qv[tid] = q[b * Ac + tid];
  __syncthreads();

  { // score for s = tid
    const float* ep = encp + ((size_t)b * Sc + tid) * Ac;
    float e = 0.f;
#pragma unroll 4
    for (int a = 0; a < Ac; ++a) e += tanhf(qv[a] + ep[a]) * Va[a];
    sc[tid] = e + bV[0];
  }
  __syncthreads();

  // max-reduce
  red[tid] = sc[tid];
  __syncthreads();
  for (int off = 128; off > 0; off >>= 1) {
    if (tid < off) red[tid] = fmaxf(red[tid], red[tid + off]);
    __syncthreads();
  }
  const float mx = red[0];
  __syncthreads();

  const float ev = __expf(sc[tid] - mx);
  red[tid] = ev;
  __syncthreads();
  for (int off = 128; off > 0; off >>= 1) {
    if (tid < off) red[tid] += red[tid + off];
    __syncthreads();
  }
  const float inv = 1.0f / red[0];
  __syncthreads();
  sc[tid] = ev * inv;                       // attn weights
  __syncthreads();

  // ctx[h] = sum_s attn[s] * enc[b,s,h]
  float* dst = act1 + (size_t)b * K1c;
  const float* eb = enc + (size_t)b * Sc * Hec;
  for (int h = tid; h < Hec; h += 256) {
    float a0 = 0.f;
    for (int s = 0; s < Sc; ++s) a0 += sc[s] * eb[(size_t)s * Hec + h];
    dst[Ec + h] = a0;
  }
  // embedding gather into act1[:, 0:E]
  const int xt = x[b * Tc + t];
  for (int i = tid; i < Ec; i += 256) dst[i] = emb[(size_t)xt * Ec + i];
}

// ---------------------------------------------------------------------------
// LSTM gate kernels (keras order i,f,g,o)
// ---------------------------------------------------------------------------
__global__ __launch_bounds__(256) void gates1_kernel(
    const float* __restrict__ z, float* __restrict__ c1,
    float* __restrict__ act1, float* __restrict__ act2)
{
  const int idx = blockIdx.x * 256 + threadIdx.x;     // B*H
  const int b = idx >> 10, j = idx & 1023;
  const float* zr = z + (size_t)b * (4 * Hc);
  const float ig = sigm(zr[j]);
  const float fg = sigm(zr[Hc + j]);
  const float gg = tanhf(zr[2 * Hc + j]);
  const float og = sigm(zr[3 * Hc + j]);
  const float c  = fg * c1[idx] + ig * gg;
  const float h  = og * tanhf(c);
  c1[idx] = c;
  act1[(size_t)b * K1c + (Ec + Hec) + j] = h;          // h1 slot (next-step q + Wh1 input)
  act2[(size_t)b * K2c + j]              = h;          // LSTM2 x-input
}

__global__ __launch_bounds__(256) void gates2_kernel(
    const float* __restrict__ z, float* __restrict__ c2,
    float* __restrict__ act2, float* __restrict__ outs, int t)
{
  const int idx = blockIdx.x * 256 + threadIdx.x;
  const int b = idx >> 10, j = idx & 1023;
  const float* zr = z + (size_t)b * (4 * Hc);
  const float ig = sigm(zr[j]);
  const float fg = sigm(zr[Hc + j]);
  const float gg = tanhf(zr[2 * Hc + j]);
  const float og = sigm(zr[3 * Hc + j]);
  const float c  = fg * c2[idx] + ig * gg;
  const float h  = og * tanhf(c);
  c2[idx] = c;
  act2[(size_t)b * K2c + Hc + j] = h;                  // h2 recurrent slot
  outs[((size_t)b * Tc + t) * Hc + j] = h;             // row b*T+t for final GEMM
}

// ---------------------------------------------------------------------------
// Utility kernels
// ---------------------------------------------------------------------------
__global__ void convert_bf16_kernel(const float* __restrict__ src, u16* __restrict__ dst, int n) {
  const int i = blockIdx.x * 256 + threadIdx.x;
  if (i < n) dst[i] = f2bf(src[i]);
}
__global__ void zero_kernel(float* __restrict__ p, int n) {
  const int i = blockIdx.x * 256 + threadIdx.x;
  if (i < n) p[i] = 0.f;
}

// ---------------------------------------------------------------------------
// Host-side orchestration
// ---------------------------------------------------------------------------
extern "C" void kernel_launch(void* const* d_in, const int* in_sizes, int n_in,
                              void* d_out, int out_size, void* d_ws, size_t ws_size,
                              hipStream_t stream) {
  (void)in_sizes; (void)n_in; (void)out_size; (void)ws_size;

  const int*   x    = (const int*)  d_in[0];
  const float* enc  = (const float*)d_in[1];
  const float* emb  = (const float*)d_in[2];
  const float* W1   = (const float*)d_in[3];
  const float* b1   = (const float*)d_in[4];
  const float* W2   = (const float*)d_in[5];
  const float* b2   = (const float*)d_in[6];
  const float* Va   = (const float*)d_in[7];
  const float* bV   = (const float*)d_in[8];
  const float* Wx1  = (const float*)d_in[9];
  const float* Wh1  = (const float*)d_in[10];
  const float* bl1  = (const float*)d_in[11];
  const float* Wx2  = (const float*)d_in[12];
  const float* Wh2  = (const float*)d_in[13];
  const float* bl2  = (const float*)d_in[14];
  const float* Wf   = (const float*)d_in[15];
  const float* bfb  = (const float*)d_in[16];

  char* ws = (char*)d_ws;
  size_t off = 0;
  auto alloc = [&](size_t bytes) -> void* {
    void* p = ws + off;
    off += (bytes + 255) & ~(size_t)255;
    return p;
  };

  u16*   Wcat1 = (u16*)  alloc((size_t)K1c * 4 * Hc * 2);     // [2304,4096] bf16
  u16*   Wcat2 = (u16*)  alloc((size_t)K2c * 4 * Hc * 2);     // [2048,4096] bf16
  u16*   W1b   = (u16*)  alloc((size_t)Hc * Ac * 2);
  u16*   W2b   = (u16*)  alloc((size_t)Hec * Ac * 2);
  u16*   Wfb   = (u16*)  alloc((size_t)Hc * Vc * 2);
  float* encp  = (float*)alloc((size_t)Bc * Sc * Ac * 4);     // enc_proj
  float* act1  = (float*)alloc((size_t)Bc * K1c * 4);         // [emb|ctx|h1]
  float* act2  = (float*)alloc((size_t)Bc * K2c * 4);         // [h1|h2]
  float* zbuf  = (float*)alloc((size_t)Bc * 4 * Hc * 4);
  float* c1    = (float*)alloc((size_t)Bc * Hc * 4);
  float* c2    = (float*)alloc((size_t)Bc * Hc * 4);
  float* qbuf  = (float*)alloc((size_t)Bc * Ac * 4);
  float* outs  = (float*)alloc((size_t)Bc * Tc * Hc * 4);

  auto cvt = [&](const float* s, u16* d, int n) {
    convert_bf16_kernel<<<(n + 255) / 256, 256, 0, stream>>>(s, d, n);
  };
  auto zero = [&](float* p, int n) {
    zero_kernel<<<(n + 255) / 256, 256, 0, stream>>>(p, n);
  };
  auto gemm = [&](const float* Aa, int lda, const u16* Bb, const float* bias,
                  float* Cc, int ldc, int M, int N, int K) {
    dim3 grid(N / BN, M / BM);
    gemm_bf16_kernel<<<grid, 256, 0, stream>>>(Aa, lda, Bb, bias, Cc, ldc, M, N, K);
  };

  // --- one-time setup ---
  zero(act1, Bc * K1c);
  zero(act2, Bc * K2c);
  zero(c1, Bc * Hc);
  zero(c2, Bc * Hc);

  cvt(Wx1, Wcat1,                        (Ec + Hec) * 4 * Hc);   // rows 0..1279
  cvt(Wh1, Wcat1 + (size_t)(Ec + Hec) * 4 * Hc, Hc * 4 * Hc);    // rows 1280..2303
  cvt(Wx2, Wcat2,                        Hc * 4 * Hc);
  cvt(Wh2, Wcat2 + (size_t)Hc * 4 * Hc,  Hc * 4 * Hc);
  cvt(W1,  W1b, Hc * Ac);
  cvt(W2,  W2b, Hec * Ac);
  cvt(Wf,  Wfb, Hc * Vc);

  // enc_proj = encoder_outputs @ W2 + b2  : [B*S, A]
  gemm(enc, Hec, W2b, b2, encp, Ac, Bc * Sc, Ac, Hec);

  // --- recurrent loop ---
  for (int t = 0; t < Tc; ++t) {
    // q = h1 @ W1 + b1 (h1 lives in act1 h-slot)
    gemm(act1 + (Ec + Hec), K1c, W1b, b1, qbuf, Ac, Bc, Ac, Hc);
    // attention + ctx + embedding gather
    attention_kernel<<<Bc, 256, 0, stream>>>(qbuf, encp, enc, Va, bV, emb, x, t, act1);
    // z1 = [emb|ctx|h1] @ Wcat1 + bl1
    gemm(act1, K1c, Wcat1, bl1, zbuf, 4 * Hc, Bc, 4 * Hc, K1c);
    gates1_kernel<<<(Bc * Hc) / 256, 256, 0, stream>>>(zbuf, c1, act1, act2);
    // z2 = [h1|h2] @ Wcat2 + bl2
    gemm(act2, K2c, Wcat2, bl2, zbuf, 4 * Hc, Bc, 4 * Hc, K2c);
    gates2_kernel<<<(Bc * Hc) / 256, 256, 0, stream>>>(zbuf, c2, act2, outs, t);
  }

  // logits = outs @ Wf + bf : [B*T, V] -> d_out
  gemm(outs, Hc, Wfb, bfb, (float*)d_out, Vc, Bc * Tc, Vc, Hc);
}
